// MultiHeadAttention_87076166959377
// MI455X (gfx1250) — matrix operand order repaired and checked by reading, not compiled, not run on hardware
//
#include <hip/hip_runtime.h>
#include <hip/hip_bf16.h>

// MI455X (gfx1250) multi-head attention, bf16 WMMA pipeline.
// GEMMs: v_wmma_f32_16x16x32_bf16; staging: global_load_async_to_lds_b128
// (ASYNCcnt) with double-buffered LDS tiles for cross-tile latency hiding.

typedef __attribute__((ext_vector_type(16))) __bf16 v16bf;
typedef __attribute__((ext_vector_type(8)))  float  v8f;

union V16 { v16bf v; uint4 q[2]; };

__device__ inline __bf16 f2bf(float f) {
  union { float f; unsigned u; } x; x.f = f;
  unsigned r = (x.u + 0x7FFFu + ((x.u >> 16) & 1u)) >> 16;   // RNE
  union { unsigned short s; __bf16 b; } y; y.s = (unsigned short)r;
  return y.b;
}

// Low 32 bits of a generic pointer to LDS == wave-relative LDS byte offset.
__device__ inline unsigned lds_off(const void* p) {
  return (unsigned)(unsigned long long)p;
}
// CDNA5 async global->LDS 16-byte copy (ASYNCcnt tracked, no VGPR data).
__device__ inline void async_b128(unsigned lds, const void* g) {
  asm volatile("global_load_async_to_lds_b128 %0, %1, off"
               :: "v"(lds), "v"((unsigned long long)g) : "memory");
}
__device__ inline void wait_async0() {
  asm volatile("s_wait_asynccnt 0x0" ::: "memory");
}

// ---------------------------------------------------------------------------
// Vectorized f32 -> bf16 conversion (float4 in, 4x bf16 out)
// ---------------------------------------------------------------------------
__global__ __launch_bounds__(256)
void cvt_f32_to_bf16(const float4* __restrict__ src, uint2* __restrict__ dst, int n4) {
  int i = blockIdx.x * 256 + threadIdx.x;
  if (i < n4) {
    float4 v = src[i];
    union { uint2 u; __bf16 h[4]; } o;
    o.h[0] = f2bf(v.x); o.h[1] = f2bf(v.y); o.h[2] = f2bf(v.z); o.h[3] = f2bf(v.w);
    dst[i] = o.u;
  }
}

// ---------------------------------------------------------------------------
// Generic batched bf16 GEMM:  C[z] = A[z] * B[z] * scale (+ bias)
//   A: MxK row-major (lda). BT=0: B row-major KxN. BT=1: B[k][n]=Bm[n*ldb+k].
//   Batch offsets: off(z) = (z >> sh) * hi + (z & ((1<<sh)-1)) * lo   (elements)
//   Block tile 128x128, BK=32; 8 waves; wave tile 32x64 = 2x4 WMMA (8/iter).
//   Both LDS tiles stored K-contiguous -> every fragment is 2x ds_load_b128.
//   Double-buffered LDS (40KB): stage tile kt+1 async while computing tile kt.
// ---------------------------------------------------------------------------
template <int BT>
__global__ __launch_bounds__(256)
void gemm_bf16_wmma(const __bf16* __restrict__ A,
                    const __bf16* __restrict__ Bm,
                    const float*  __restrict__ bias,
                    float* out_f32, __bf16* out_bf16,
                    int M, int N, int K, int lda, int ldb, int ldo,
                    float scale,
                    int a_sh, long long a_hi, long long a_lo,
                    int b_sh, long long b_hi, long long b_lo,
                    int o_sh, long long o_hi, long long o_lo,
                    int bias_sh, int bias_stride)
{
  constexpr int BM = 128, BN = 128, BK = 32;
  constexpr int LD = BK + 8;                       // 40 elems -> 80B row stride
  constexpr unsigned BUFB = BM * LD * 2;           // 10240B per buffer
  __shared__ __attribute__((aligned(16))) __bf16 As[2][BM][LD];   // As[buf][m][k]
  __shared__ __attribute__((aligned(16))) __bf16 Bt[2][BN][LD];   // Bt[buf][n][k]

  const int tid  = threadIdx.x;
  const int lane = tid & 31;
  const int wave = tid >> 5;
  const int wm   = wave & 3;     // 32-row slice
  const int wn   = wave >> 2;    // 64-col slice
  const int lm   = lane & 15;
  const int lhi  = lane >> 4;

  const long long z = blockIdx.z;
  const size_t a_off = (size_t)(((z >> a_sh) * a_hi) + ((z & ((1LL << a_sh) - 1)) * a_lo));
  const size_t b_off = (size_t)(((z >> b_sh) * b_hi) + ((z & ((1LL << b_sh) - 1)) * b_lo));
  const size_t o_off = (size_t)(((z >> o_sh) * o_hi) + ((z & ((1LL << o_sh) - 1)) * o_lo));
  const int bias_off = (int)((z & ((1LL << bias_sh) - 1)) * bias_stride);

  const int m0 = blockIdx.y * BM;
  const int n0 = blockIdx.x * BN;

  // ---- per-thread staging pointers (advance by BK each K-step) ----
  const __bf16* pA[2]; unsigned lA[2];
  #pragma unroll
  for (int i = 0; i < 2; ++i) {
    int idx = tid + 256 * i;
    int r = idx >> 2, c = (idx & 3) * 8;        // 128 rows x 32 cols, 16B chunks
    pA[i] = A + a_off + (size_t)(m0 + r) * lda + c;
    lA[i] = lds_off(&As[0][r][c]);
  }
  const __bf16* pB[2]; unsigned lB[2]; int sk[2], sn[2];
  #pragma unroll
  for (int i = 0; i < 2; ++i) {
    int idx = tid + 256 * i;
    if (BT) {               // source row n is K-contiguous -> direct async copy
      int n = idx >> 2, c = (idx & 3) * 8;
      pB[i] = Bm + b_off + (size_t)(n0 + n) * ldb + c;
      lB[i] = lds_off(&Bt[0][n][c]);
      sk[i] = 0; sn[i] = 0;
    } else {                // row-major KxN source -> transpose-scatter into Bt
      int k = idx >> 4, ng = (idx & 15) * 8;    // 32 k-rows x 128 n, 8 n per thread
      pB[i] = Bm + b_off + (size_t)k * ldb + n0 + ng;
      lB[i] = 0; sk[i] = k; sn[i] = ng;
    }
  }

  // Issue all copies for one K-tile into LDS buffer `buf`, then advance ptrs.
  auto stage = [&](int buf) {
    const unsigned o = (unsigned)buf * BUFB;
    async_b128(lA[0] + o, pA[0]);
    async_b128(lA[1] + o, pA[1]);
    if (BT) {
      async_b128(lB[0] + o, pB[0]);
      async_b128(lB[1] + o, pB[1]);
      pB[0] += BK; pB[1] += BK;
    } else {
      #pragma unroll
      for (int i = 0; i < 2; ++i) {
        union { uint4 q; __bf16 h[8]; } t;
        t.q = *reinterpret_cast<const uint4*>(pB[i]);
        #pragma unroll
        for (int j = 0; j < 8; ++j) Bt[buf][sn[i] + j][sk[i]] = t.h[j];
      }
      pB[0] += (size_t)BK * ldb; pB[1] += (size_t)BK * ldb;
    }
    pA[0] += BK; pA[1] += BK;
    __builtin_prefetch(pA[0], 0, 0);   // speculative; dropped past end of tensor
  };

  v8f zero = {0.f,0.f,0.f,0.f,0.f,0.f,0.f,0.f};
  v8f acc[2][4];
  #pragma unroll
  for (int i = 0; i < 2; ++i)
    #pragma unroll
    for (int j = 0; j < 4; ++j) acc[i][j] = zero;

  const int ktiles = K / BK;
  stage(0);                               // prologue: tile 0 -> buffer 0
  for (int kt = 0; kt < ktiles; ++kt) {
    wait_async0();                        // this wave's copies for tile kt landed
    __syncthreads();                      // all waves' copies landed
    if (kt + 1 < ktiles) stage((kt + 1) & 1);   // overlap next tile with compute
    const int cur = kt & 1;

    // ---- fragments: every load is an aligned ds_load_b128 ----
    v16bf afr[2], bfr[4];
    #pragma unroll
    for (int tm = 0; tm < 2; ++tm) {
      int mr = wm * 32 + tm * 16 + lm;
      V16 t;
      t.q[0] = *reinterpret_cast<const uint4*>(&As[cur][mr][lhi * 8]);
      t.q[1] = *reinterpret_cast<const uint4*>(&As[cur][mr][16 + lhi * 8]);
      afr[tm] = t.v;
    }
    #pragma unroll
    for (int tn = 0; tn < 4; ++tn) {
      int nc = wn * 64 + tn * 16 + lm;
      V16 t;
      t.q[0] = *reinterpret_cast<const uint4*>(&Bt[cur][nc][lhi * 8]);
      t.q[1] = *reinterpret_cast<const uint4*>(&Bt[cur][nc][16 + lhi * 8]);
      bfr[tn] = t.v;
    }
    #pragma unroll
    for (int tm = 0; tm < 2; ++tm)
      #pragma unroll
      for (int tn = 0; tn < 4; ++tn)
        acc[tm][tn] = __builtin_amdgcn_wmma_f32_16x16x32_bf16(
            false, afr[tm], false, bfr[tn], (short)0, acc[tm][tn], false, false);
    __syncthreads();                      // done reading buf `cur`
  }

  // ---- epilogue: scale, bias, store f32 or bf16 ----
  #pragma unroll
  for (int tm = 0; tm < 2; ++tm) {
    #pragma unroll
    for (int tn = 0; tn < 4; ++tn) {
      #pragma unroll
      for (int r = 0; r < 8; ++r) {
        int m = m0 + wm * 32 + tm * 16 + r + 8 * lhi;
        int n = n0 + wn * 64 + tn * 16 + lm;
        float v = acc[tm][tn][r] * scale;
        if (bias) v += bias[bias_off + n];
        size_t oi = o_off + (size_t)m * ldo + n;
        if (out_f32) out_f32[oi] = v;
        else         out_bf16[oi] = f2bf(v);
      }
    }
  }
}

// ---------------------------------------------------------------------------
// Row softmax: one 256-thread block per row of 1024 f32 scores.
// Writes bf16 probs in-place into the upper half of the f32 row
// (bf16 index = row*2048 + 1024 + t). Safe: all reads precede first barrier.
// ---------------------------------------------------------------------------
__global__ __launch_bounds__(256)
void softmax_rows(const float* __restrict__ scores, __bf16* attn_base) {
  const int row = blockIdx.x;
  const int tid = threadIdx.x;
  const float4* sp = reinterpret_cast<const float4*>(scores + (size_t)row * 1024);
  __shared__ float red[256];

  float4 v4 = sp[tid];
  float v[4] = {v4.x, v4.y, v4.z, v4.w};
  float lmax = fmaxf(fmaxf(v[0], v[1]), fmaxf(v[2], v[3]));
  red[tid] = lmax; __syncthreads();
  for (int s = 128; s > 0; s >>= 1) {
    if (tid < s) red[tid] = fmaxf(red[tid], red[tid + s]);
    __syncthreads();
  }
  const float mx = red[0]; __syncthreads();

  float e[4]; float lsum = 0.f;
  #pragma unroll
  for (int i = 0; i < 4; ++i) { e[i] = __expf(v[i] - mx); lsum += e[i]; }
  red[tid] = lsum; __syncthreads();
  for (int s = 128; s > 0; s >>= 1) {
    if (tid < s) red[tid] += red[tid + s];
    __syncthreads();
  }
  const float inv = 1.f / red[0];

  union { uint2 u; __bf16 h[4]; } o;
  #pragma unroll
  for (int i = 0; i < 4; ++i) o.h[i] = f2bf(e[i] * inv);
  uint2* ap = reinterpret_cast<uint2*>(attn_base + (size_t)row * 2048 + 1024);
  ap[tid] = o.u;
}

// ---------------------------------------------------------------------------
// Host launch
// ---------------------------------------------------------------------------
extern "C" void kernel_launch(void* const* d_in, const int* in_sizes, int n_in,
                              void* d_out, int out_size, void* d_ws, size_t ws_size,
                              hipStream_t stream) {
  (void)in_sizes; (void)n_in; (void)out_size; (void)ws_size;
  const float* x  = (const float*)d_in[0];
  const float* Wq = (const float*)d_in[1];
  const float* Wk = (const float*)d_in[2];
  const float* Wv = (const float*)d_in[3];
  const float* bq = (const float*)d_in[4];
  const float* bk = (const float*)d_in[5];
  const float* bv = (const float*)d_in[6];
  const float* Wo = (const float*)d_in[7];
  const float* bo = (const float*)d_in[8];
  float* out = (float*)d_out;

  constexpr long long Bz = 8, S = 1024, E = 512, H = 8;
  constexpr long long XN = Bz * S * E;        // 4,194,304
  constexpr long long WN = H * E * E;         // 2,097,152
  constexpr long long QN = Bz * H * S * E;    // 33,554,432
  // Workspace (bf16 elems unless noted); ~495 MB total.
  __bf16* xb   = (__bf16*)d_ws;
  __bf16* wqb  = xb  + XN;
  __bf16* wkb  = wqb + WN;
  __bf16* wvb  = wkb + WN;
  __bf16* wob  = wvb + WN;
  __bf16* qb   = wob + WN;        // reused as concat buffer after scores GEMM
  __bf16* kb   = qb  + QN;
  __bf16* vb   = kb  + QN;
  float*  sc   = (float*)(vb + QN);       // [B*H, 1024, 1024] f32 scores
  __bf16* attn = (__bf16*)sc;             // aliased: bf16 probs in upper half of rows
  __bf16* concat = qb;

  // 1) convert inputs to bf16 (float4 vectorized)
  cvt_f32_to_bf16<<<(int)(XN / 1024), 256, 0, stream>>>((const float4*)x,  (uint2*)xb,  (int)(XN / 4));
  cvt_f32_to_bf16<<<(int)(WN / 1024), 256, 0, stream>>>((const float4*)Wq, (uint2*)wqb, (int)(WN / 4));
  cvt_f32_to_bf16<<<(int)(WN / 1024), 256, 0, stream>>>((const float4*)Wk, (uint2*)wkb, (int)(WN / 4));
  cvt_f32_to_bf16<<<(int)(WN / 1024), 256, 0, stream>>>((const float4*)Wv, (uint2*)wvb, (int)(WN / 4));
  cvt_f32_to_bf16<<<(int)(WN / 1024), 256, 0, stream>>>((const float4*)Wo, (uint2*)wob, (int)(WN / 4));

  // 2) Q/K/V projections: z = b*H + h ; A=x[b] (shared over h), B=W[h]
  dim3 gp(E / 128, S / 128, Bz * H);
  gemm_bf16_wmma<0><<<gp, 256, 0, stream>>>(xb, wqb, bq, nullptr, qb,
      (int)S, (int)E, (int)E, (int)E, (int)E, (int)E, 1.0f,
      3, S * E, 0,   3, 0, E * E,   0, S * E, 0,   3, (int)E);
  gemm_bf16_wmma<0><<<gp, 256, 0, stream>>>(xb, wkb, bk, nullptr, kb,
      (int)S, (int)E, (int)E, (int)E, (int)E, (int)E, 1.0f,
      3, S * E, 0,   3, 0, E * E,   0, S * E, 0,   3, (int)E);
  gemm_bf16_wmma<0><<<gp, 256, 0, stream>>>(xb, wvb, bv, nullptr, vb,
      (int)S, (int)E, (int)E, (int)E, (int)E, (int)E, 1.0f,
      3, S * E, 0,   3, 0, E * E,   0, S * E, 0,   3, (int)E);

  // 3) scores = q @ k^T * 1/sqrt(E)   (f32 out; BT source is K-contiguous)
  dim3 gs(S / 128, S / 128, Bz * H);
  gemm_bf16_wmma<1><<<gs, 256, 0, stream>>>(qb, kb, nullptr, sc, nullptr,
      (int)S, (int)S, (int)E, (int)E, (int)E, (int)S, 0.04419417382415922f,
      0, S * E, 0,   0, S * E, 0,   0, S * S, 0,   0, 0);

  // 4) softmax over each row of 1024
  softmax_rows<<<(int)(Bz * H * S), 256, 0, stream>>>(sc, attn);

  // 5) o = attn @ v, written directly in concat layout [B, S, H*E]
  dim3 ga(E / 128, S / 128, Bz * H);
  gemm_bf16_wmma<0><<<ga, 256, 0, stream>>>(attn + 1024, vb, nullptr, nullptr, concat,
      (int)S, (int)E, (int)S, 2048, (int)E, (int)(H * E), 1.0f,
      0, 2 * S * S, 0,   0, S * E, 0,   3, S * H * E, E,   0, 0);

  // 6) out = concat @ Wo + bo   (f32 out)
  dim3 go(E / 128, (Bz * S) / 128, 1);
  gemm_bf16_wmma<0><<<go, 256, 0, stream>>>(concat, wob, bo, out, nullptr,
      (int)(Bz * S), (int)E, (int)(H * E), (int)(H * E), (int)E, (int)E, 1.0f,
      0, 0, 0,   0, 0, 0,   0, 0, 0,   0, 0);
}
